// VoxelBackBone8x_78984448573936
// MI455X (gfx1250) — matrix operand order repaired and checked by reading, not compiled, not run on hardware
//
#include <hip/hip_runtime.h>
#include <hip/hip_bf16.h>
#include <stdint.h>
#include <stddef.h>

typedef __attribute__((ext_vector_type(16))) __bf16 v16bf;
typedef __attribute__((ext_vector_type(8)))  float  v8f;
typedef __attribute__((ext_vector_type(4)))  unsigned int v4u;
typedef __attribute__((ext_vector_type(8)))  int v8i;
typedef __attribute__((ext_vector_type(4)))  int v4i;
typedef unsigned short u16s;
typedef unsigned int   u32s;

struct __align__(16) q128 { u32s x, y, z, w; };

union AFrag {
  v16bf v;
  q128  q[2];
};

__device__ __forceinline__ u16s f2bf(float f) {
  u32s u = __float_as_uint(f);
  u32s r = ((u >> 16) & 1u) + 0x7FFFu;   // round-to-nearest-even
  return (u16s)((u + r) >> 16);
}

// ---------------------------------------------------------------------------
// Scatter voxel features into dense 4-channel grid + occupancy mask
// ---------------------------------------------------------------------------
__global__ __launch_bounds__(256) void scatter_voxels(
    const float* __restrict__ feats, const int* __restrict__ coords,
    float* __restrict__ dense4, float* __restrict__ mask0,
    int n, int H, int W)
{
  int i = blockIdx.x * 256 + threadIdx.x;
  if (i >= n) return;
  int z = coords[i * 4 + 1];
  int y = coords[i * 4 + 2];
  int x = coords[i * 4 + 3];
  size_t vox = ((size_t)z * H + y) * W + x;
  for (int c = 0; c < 4; ++c)
    atomicAdd(&dense4[vox * 4 + c], feats[i * 4 + c]);
  mask0[vox] = 1.0f;
}

// ---------------------------------------------------------------------------
// f32 4-ch dense -> bf16 16-ch (zero padded) NDHWC
// ---------------------------------------------------------------------------
__global__ __launch_bounds__(256) void pad_to_bf16(
    const float* __restrict__ dense4, u16s* __restrict__ act, size_t nv)
{
  size_t t = (size_t)blockIdx.x * 256 + threadIdx.x;
  if (t >= nv * 16) return;
  size_t vox = t >> 4;
  int c = (int)(t & 15);
  float v = (c < 4) ? dense4[vox * 4 + c] : 0.0f;
  act[t] = f2bf(v);
}

// ---------------------------------------------------------------------------
// Pack fp32 (KD,KH,KW,CIr,COUT) weights into bf16 WMMA B-fragments.
// K flattened PER kd-PLANE, padded to 32. Global chunk q = kd*PCH + pq.
// B layout (16x16x32 bf16): lane holds column N = lane%16; element e holds
// K = q*32 + (e/8)*16 + (lane/16)*8 + e%8.
// ---------------------------------------------------------------------------
__global__ __launch_bounds__(256) void pack_weights(
    const float* __restrict__ w, u16s* __restrict__ out,
    int KD, int KH, int KW, int CIr, int CIN, int COUT, int PCH)
{
  int t = blockIdx.x * 256 + threadIdx.x;
  int NT = COUT >> 4;
  int NCH = KD * PCH;
  int total = NCH * NT * 512;
  if (t >= total) return;
  int e    = t & 15;
  int lane = (t >> 4) & 31;
  int f    = t >> 9;
  int q    = f / NT;
  int n    = f - q * NT;
  int half = lane >> 4;
  int nc   = n * 16 + (lane & 15);
  int kd   = q / PCH;
  int pq   = q - kd * PCH;
  int kp   = pq * 32 + (e >> 3) * 16 + half * 8 + (e & 7);      // K within the kd-plane (padded)
  int PK   = KH * KW * CIN;
  float val = 0.0f;
  if (kp < PK) {
    int tip = kp / CIN;                    // tap in plane
    int c   = kp - tip * CIN;
    int kh  = tip / KW;
    int kw  = tip - kh * KW;
    if (c < CIr)
      val = w[(size_t)(((kd * KH + kh) * KW + kw) * CIr + c) * COUT + nc];
  }
  out[t] = f2bf(val);
}

// ---------------------------------------------------------------------------
// Fold BN into per-channel scale/bias: y = x*s + b; s = g*rsqrt(v+eps)
// ---------------------------------------------------------------------------
__global__ __launch_bounds__(256) void fold_bn(
    const float* __restrict__ g, const float* __restrict__ b,
    const float* __restrict__ m, const float* __restrict__ v,
    float* __restrict__ sc, float* __restrict__ bi, int C)
{
  int c = blockIdx.x * 256 + threadIdx.x;
  if (c >= C) return;
  float s = g[c] * rsqrtf(v[c] + 1e-3f);
  sc[c] = s;
  bi[c] = b[c] - m[c] * s;
}

// ---------------------------------------------------------------------------
// Mask dilation (binary "any in window")
// ---------------------------------------------------------------------------
template <int KD, int KH, int KW, int SD, int SH, int SW, int PD, int PH, int PW>
__global__ __launch_bounds__(256) void dilate_mask(
    const float* __restrict__ mi, float* __restrict__ mo,
    int DI, int HI, int WI, int DO, int HO, int WO)
{
  int t = blockIdx.x * 256 + threadIdx.x;
  int tot = DO * HO * WO;
  if (t >= tot) return;
  int od = t / (HO * WO);
  int r  = t - od * HO * WO;
  int oh = r / WO;
  int ow = r - oh * WO;
  float acc = 0.0f;
  for (int kd = 0; kd < KD; ++kd) {
    int id = od * SD - PD + kd;
    if (id < 0 || id >= DI) continue;
    for (int kh = 0; kh < KH; ++kh) {
      int ih = oh * SH - PH + kh;
      if (ih < 0 || ih >= HI) continue;
      for (int kw = 0; kw < KW; ++kw) {
        int iw = ow * SW - PW + kw;
        if (iw < 0 || iw >= WI) continue;
        acc += mi[((size_t)id * HI + ih) * WI + iw];
      }
    }
  }
  mo[t] = (acc > 0.0f) ? 1.0f : 0.0f;
}

// ---------------------------------------------------------------------------
// A-fragment loader: pq is always an unrolled constant at call sites, so the
// /CIN, /KW, %KW fold; address = scalar rowoff[kh] + per-lane laneoff + imm.
// ---------------------------------------------------------------------------
template <int CIN, int KH, int KW, int PK>
__device__ __forceinline__ AFrag load_afrag(
    const u16s* __restrict__ in, const long* rowoff,
    const bool* hok, const bool* wok, int laneoff, int pq)
{
  AFrag a;
#pragma unroll
  for (int g = 0; g < 2; ++g) {
    const int kp = pq * 32 + g * 16;
    q128 dat; dat.x = 0u; dat.y = 0u; dat.z = 0u; dat.w = 0u;
    if (kp < PK) {
      const int tip = kp / CIN;
      const int kh  = tip / KW;
      const int kw  = tip - kh * KW;
      const int cc  = kp % CIN;
      if (hok[kh] && wok[kw]) {
        dat = *(const q128*)(in + rowoff[kh] + laneoff + (kw * CIN + cc));
      }
    }
    a.q[g] = dat;
  }
  return a;
}

// ---------------------------------------------------------------------------
// Implicit-GEMM 3D conv + folded BN + ReLU + mask, bf16 WMMA, f32 accum.
// One wave computes 16 W-contiguous voxels x COUT. Tile index is forced
// wave-uniform (readfirstlane) so tile/row math lives in SGPRs; A fragments
// are software-pipelined one chunk ahead. Weights staged to LDS by the TDM.
// ---------------------------------------------------------------------------
template <int CIN, int COUT, int KD, int KH, int KW,
          int SD, int SH, int SW, int PD, int PH, int PW,
          bool USE_LDS, bool OUTF32>
__global__ __launch_bounds__(256) void conv_bn_relu_wmma(
    const u16s* __restrict__ in, const u16s* __restrict__ wpk,
    const float* __restrict__ scale, const float* __restrict__ bias,
    const float* __restrict__ maskv,
    u16s* __restrict__ outb, float* __restrict__ outf,
    int DI, int HI, int WI, int DO, int HO, int WO)
{
  constexpr int NT    = COUT / 16;
  constexpr int PTAPS = KH * KW;                // taps per kd-plane
  constexpr int PK    = PTAPS * CIN;            // real K per plane
  constexpr int PCH   = (PK + 31) / 32;         // padded chunks per plane

  extern __shared__ u16s sW[];
  const int tid  = threadIdx.x;
  const int lane = tid & 31;
  const int wv   = tid >> 5;
  const int half = lane >> 4;
  const int m16  = lane & 15;

  if (USE_LDS) {
    if (wv == 0) {
      // Tensor Data Mover: DMA the whole packed-weight buffer into LDS.
      constexpr unsigned NBYTES = (unsigned)(KD * PCH) * NT * 1024u;
      constexpr unsigned N8 = NBYTES / 8u;
      unsigned long long ga = (unsigned long long)(size_t)wpk;
      unsigned ldsoff = (unsigned)(size_t)(void*)sW;
      v4u g0;
      g0[0] = 1u;                                   // count=1, user mode
      g0[1] = ldsoff;                               // lds_addr
      g0[2] = (unsigned)ga;                         // global_addr[31:0]
      g0[3] = (unsigned)(ga >> 32) | 0x80000000u;   // global_addr[56:32] | type=2
      v8i g1;
      g1[0] = (int)(3u << 16);                      // data_size = 8B
      g1[1] = (int)((N8 & 0xFFFFu) << 16);          // tensor_dim0[15:0]
      g1[2] = (int)(((N8 >> 16) & 0xFFFFu) | (1u << 16)); // dim0 hi | tensor_dim1=1
      g1[3] = (int)((N8 & 0xFFFFu) << 16);          // tile_dim0
      g1[4] = 1;                                    // tile_dim1 = 1
      g1[5] = (int)N8;                              // tensor_dim0_stride[31:0]
      g1[6] = 0;
      g1[7] = 0;
      v4i g2; g2[0] = 0; g2[1] = 0; g2[2] = 0; g2[3] = 0;
      v4i g3; g3[0] = 0; g3[1] = 0; g3[2] = 0; g3[3] = 0;
      v8i g4; g4[0] = 0; g4[1] = 0; g4[2] = 0; g4[3] = 0;
      g4[4] = 0; g4[5] = 0; g4[6] = 0; g4[7] = 0;
      __builtin_amdgcn_tensor_load_to_lds(g0, g1, g2, g3, g4, 0);
      __builtin_amdgcn_s_wait_tensorcnt(0);
    }
    __syncthreads();
  }

  const int tilesW = WO >> 4;                 // WO is a multiple of 16 here
  const int totalTiles = DO * HO * tilesW;
  // Force wave-uniform tile index into SGPRs.
  const int tile = __builtin_amdgcn_readfirstlane(blockIdx.x * 8 + wv);
  if (tile >= totalTiles) return;

  const int od  = tile / (HO * tilesW);
  int rem       = tile - od * (HO * tilesW);
  const int oh  = rem / tilesW;
  const int ow0 = (rem - oh * tilesW) << 4;

  const int idb = od * SD - PD;
  const int ihb = oh * SH - PH;
  const int iwb = (ow0 + m16) * SW - PW;      // per-lane input column base
  const int laneoff = iwb * CIN + half * 8;   // per-lane element offset

  bool wok[KW];
#pragma unroll
  for (int kw = 0; kw < KW; ++kw) {
    const int iw = iwb + kw;
    wok[kw] = (iw >= 0) && (iw < WI);
  }
  bool hok[KH];
#pragma unroll
  for (int kh = 0; kh < KH; ++kh) {
    const int ih = ihb + kh;
    hok[kh] = (ih >= 0) && (ih < HI);
  }

  v8f acc[NT];
  v8f zero = {};
#pragma unroll
  for (int n = 0; n < NT; ++n) acc[n] = zero;

#pragma unroll
  for (int kd = 0; kd < KD; ++kd) {
    const int id = idb + kd;
    if (id < 0 || id >= DI) continue;         // wave-uniform plane skip
    long rowoff[KH];
#pragma unroll
    for (int kh = 0; kh < KH; ++kh)
      rowoff[kh] = (long)(id * HI + (ihb + kh)) * WI * CIN;

    // Software-pipelined A fragments: load chunk pq+1 while MMAing chunk pq.
    AFrag acur = load_afrag<CIN, KH, KW, PK>(in, rowoff, hok, wok, laneoff, 0);
#pragma unroll
    for (int pq = 0; pq < PCH; ++pq) {
      AFrag anext;
      if (pq + 1 < PCH) {
        anext = load_afrag<CIN, KH, KW, PK>(in, rowoff, hok, wok, laneoff, pq + 1);
      } else {
        anext.q[0] = acur.q[0];               // dead value, never used
        anext.q[1] = acur.q[1];
      }
#pragma unroll
      for (int n = 0; n < NT; ++n) {
        AFrag bfr;
        const int base = (((kd * PCH + pq) * NT + n) * 32 + lane) * 16;
        const q128* bp = USE_LDS ? (const q128*)(sW + base)
                                 : (const q128*)(wpk + base);
        bfr.q[0] = bp[0];
        bfr.q[1] = bp[1];
        acc[n] = __builtin_amdgcn_wmma_f32_16x16x32_bf16(
            false, acur.v, false, bfr.v, (short)0, acc[n], false, false);
      }
      acur = anext;
    }
  }

  // Epilogue: C/D layout -> VGPR i holds row M = i + 8*half, col N = lane%16.
  const size_t voxbase = (size_t)(od * HO + oh) * WO + ow0;
  float mrow[8];
#pragma unroll
  for (int i = 0; i < 8; ++i)
    mrow[i] = maskv[voxbase + i + half * 8];

#pragma unroll
  for (int n = 0; n < NT; ++n) {
    const int nc = n * 16 + m16;
    const float sc = scale[nc];
    const float bi = bias[nc];
#pragma unroll
    for (int i = 0; i < 8; ++i) {
      const size_t vox = voxbase + i + half * 8;
      float val = acc[n][i] * sc + bi;
      val = val > 0.0f ? val : 0.0f;
      val *= mrow[i];
      if (OUTF32) outf[vox * COUT + nc] = val;
      else        outb[vox * COUT + nc] = f2bf(val);
    }
  }
}

// ---------------------------------------------------------------------------
// Host-side orchestration
// ---------------------------------------------------------------------------
extern "C" void kernel_launch(void* const* d_in, const int* in_sizes, int n_in,
                              void* d_out, int out_size, void* d_ws, size_t ws_size,
                              hipStream_t stream)
{
  (void)in_sizes; (void)n_in; (void)out_size; (void)ws_size;

  const float* feats  = (const float*)d_in[0];
  const int*   coords = (const int*)d_in[1];

  // params flattened in dict order: 12 x (w, bn_g, bn_b, bn_m, bn_v)
  const float* W[12]; const float* Bg[12]; const float* Bb[12];
  const float* Bm[12]; const float* Bv[12];
  for (int i = 0; i < 12; ++i) {
    W[i]  = (const float*)d_in[3 + i * 5];
    Bg[i] = (const float*)d_in[4 + i * 5];
    Bb[i] = (const float*)d_in[5 + i * 5];
    Bm[i] = (const float*)d_in[6 + i * 5];
    Bv[i] = (const float*)d_in[7 + i * 5];
  }

  // Level dims
  const int D0 = 41, H0 = 256, W0 = 256;
  const int D1 = 21, H1 = 128, W1 = 128;
  const int D2 = 11, H2 = 64,  W2 = 64;
  const int D3 = 5,  H3 = 32,  W3 = 32;
  const int D4 = 2,  H4 = 32,  W4 = 32;
  const size_t NV0 = (size_t)D0 * H0 * W0;
  const size_t NV1 = (size_t)D1 * H1 * W1;
  const size_t NV2 = (size_t)D2 * H2 * W2;
  const size_t NV3 = (size_t)D3 * H3 * W3;
  const size_t NV4 = (size_t)D4 * H4 * W4;

  // Workspace bump allocator
  char* base = (char*)d_ws;
  size_t off = 0;
  auto alloc = [&](size_t bytes) -> char* {
    char* p = base + off;
    off = (off + bytes + 255) & ~(size_t)255;
    return p;
  };

  float* dense4 = (float*)alloc(NV0 * 4 * sizeof(float));
  u16s*  actA   = (u16s*) alloc(NV0 * 16 * sizeof(u16s));
  u16s*  actB   = (u16s*) alloc(NV0 * 16 * sizeof(u16s));
  float* mask0  = (float*)alloc(NV0 * sizeof(float));
  float* mask1  = (float*)alloc(NV1 * sizeof(float));
  float* mask2  = (float*)alloc(NV2 * sizeof(float));
  float* mask3  = (float*)alloc(NV3 * sizeof(float));
  float* mask4  = (float*)alloc(NV4 * sizeof(float));

  // Layer geometry tables: KD,KH,KW, CIr, CIN, COUT
  const int LKD[12]  = {3,3,3,3,3,3,3,3,3,3,3,3};
  const int LKH[12]  = {3,3,3,3,3,3,3,3,3,3,3,1};
  const int LKW[12]  = {3,3,3,3,3,3,3,3,3,3,3,1};
  const int LCIr[12] = {4,16,16,32,32,32,64,64,64,64,64,64};
  const int LCIN[12] = {16,16,16,32,32,32,64,64,64,64,64,64};
  const int LCO[12]  = {16,16,32,32,32,64,64,64,64,64,64,128};

  u16s* pk[12]; float* sc[12]; float* bi[12];
  int   pch[12]; int nch[12];
  for (int i = 0; i < 12; ++i) {
    int PK = LKH[i] * LKW[i] * LCIN[i];
    pch[i] = (PK + 31) / 32;
    nch[i] = LKD[i] * pch[i];
    int NT = LCO[i] / 16;
    pk[i] = (u16s*)alloc((size_t)nch[i] * NT * 512 * sizeof(u16s));
    sc[i] = (float*)alloc(LCO[i] * sizeof(float));
    bi[i] = (float*)alloc(LCO[i] * sizeof(float));
  }

  // --- Stage 0: zero dense grid + mask0, scatter voxels, pad to bf16 ------
  (void)hipMemsetAsync(dense4, 0, NV0 * 4 * sizeof(float), stream);
  (void)hipMemsetAsync(mask0, 0, NV0 * sizeof(float), stream);

  {
    int n = 30000;
    scatter_voxels<<<(n + 255) / 256, 256, 0, stream>>>(
        feats, coords, dense4, mask0, n, H0, W0);
  }
  {
    size_t tot = NV0 * 16;
    pad_to_bf16<<<(unsigned)((tot + 255) / 256), 256, 0, stream>>>(
        dense4, actA, NV0);
  }

  // --- Stage 1: pack weights + fold BN ------------------------------------
  for (int i = 0; i < 12; ++i) {
    int NT = LCO[i] / 16;
    int total = nch[i] * NT * 512;
    pack_weights<<<(total + 255) / 256, 256, 0, stream>>>(
        W[i], pk[i], LKD[i], LKH[i], LKW[i], LCIr[i], LCIN[i], LCO[i], pch[i]);
    fold_bn<<<1, 256, 0, stream>>>(Bg[i], Bb[i], Bm[i], Bv[i], sc[i], bi[i], LCO[i]);
  }

  // --- Stage 2: masks ------------------------------------------------------
  dilate_mask<3,3,3,2,2,2,1,1,1><<<(unsigned)((NV1 + 255) / 256), 256, 0, stream>>>(
      mask0, mask1, D0, H0, W0, D1, H1, W1);
  dilate_mask<3,3,3,2,2,2,1,1,1><<<(unsigned)((NV2 + 255) / 256), 256, 0, stream>>>(
      mask1, mask2, D1, H1, W1, D2, H2, W2);
  dilate_mask<3,3,3,2,2,2,0,1,1><<<(unsigned)((NV3 + 255) / 256), 256, 0, stream>>>(
      mask2, mask3, D2, H2, W2, D3, H3, W3);
  dilate_mask<3,1,1,2,1,1,0,0,0><<<(unsigned)((NV4 + 255) / 256), 256, 0, stream>>>(
      mask3, mask4, D3, H3, W3, D4, H4, W4);

  // --- Stage 3: conv chain -------------------------------------------------
  auto blocksFor = [](int DO_, int HO_, int WO_) {
    int tiles = DO_ * HO_ * (WO_ >> 4);
    return (unsigned)((tiles + 7) / 8);
  };

  // LDS bytes per layer: nch*NT*1024
  // w_in : 16(pad4)->16, s1 p1, LDS 15KB (TDM staged)
  conv_bn_relu_wmma<16,16,3,3,3,1,1,1,1,1,1,true,false>
      <<<blocksFor(D0,H0,W0), 256, 15 * 1024, stream>>>(
      actA, pk[0], sc[0], bi[0], mask0, actB, nullptr, D0,H0,W0, D0,H0,W0);
  // w1 : 16->16
  conv_bn_relu_wmma<16,16,3,3,3,1,1,1,1,1,1,true,false>
      <<<blocksFor(D0,H0,W0), 256, 15 * 1024, stream>>>(
      actB, pk[1], sc[1], bi[1], mask0, actA, nullptr, D0,H0,W0, D0,H0,W0);
  // w2a : 16->32, s2 p1, LDS 30KB
  conv_bn_relu_wmma<16,32,3,3,3,2,2,2,1,1,1,true,false>
      <<<blocksFor(D1,H1,W1), 256, 30 * 1024, stream>>>(
      actA, pk[2], sc[2], bi[2], mask1, actB, nullptr, D0,H0,W0, D1,H1,W1);
  // w2b : 32->32, LDS 54KB
  conv_bn_relu_wmma<32,32,3,3,3,1,1,1,1,1,1,true,false>
      <<<blocksFor(D1,H1,W1), 256, 54 * 1024, stream>>>(
      actB, pk[3], sc[3], bi[3], mask1, actA, nullptr, D1,H1,W1, D1,H1,W1);
  // w2c : 32->32
  conv_bn_relu_wmma<32,32,3,3,3,1,1,1,1,1,1,true,false>
      <<<blocksFor(D1,H1,W1), 256, 54 * 1024, stream>>>(
      actA, pk[4], sc[4], bi[4], mask1, actB, nullptr, D1,H1,W1, D1,H1,W1);
  // w3a : 32->64, s2 p1 (weights from L2, 108KB packed)
  conv_bn_relu_wmma<32,64,3,3,3,2,2,2,1,1,1,false,false>
      <<<blocksFor(D2,H2,W2), 256, 0, stream>>>(
      actB, pk[5], sc[5], bi[5], mask2, actA, nullptr, D1,H1,W1, D2,H2,W2);
  // w3b : 64->64
  conv_bn_relu_wmma<64,64,3,3,3,1,1,1,1,1,1,false,false>
      <<<blocksFor(D2,H2,W2), 256, 0, stream>>>(
      actA, pk[6], sc[6], bi[6], mask2, actB, nullptr, D2,H2,W2, D2,H2,W2);
  // w3c : 64->64
  conv_bn_relu_wmma<64,64,3,3,3,1,1,1,1,1,1,false,false>
      <<<blocksFor(D2,H2,W2), 256, 0, stream>>>(
      actB, pk[7], sc[7], bi[7], mask2, actA, nullptr, D2,H2,W2, D2,H2,W2);
  // w4a : 64->64, s2, pad (0,1,1)
  conv_bn_relu_wmma<64,64,3,3,3,2,2,2,0,1,1,false,false>
      <<<blocksFor(D3,H3,W3), 256, 0, stream>>>(
      actA, pk[8], sc[8], bi[8], mask3, actB, nullptr, D2,H2,W2, D3,H3,W3);
  // w4b : 64->64
  conv_bn_relu_wmma<64,64,3,3,3,1,1,1,1,1,1,false,false>
      <<<blocksFor(D3,H3,W3), 256, 0, stream>>>(
      actB, pk[9], sc[9], bi[9], mask3, actA, nullptr, D3,H3,W3, D3,H3,W3);
  // w4c : 64->64
  conv_bn_relu_wmma<64,64,3,3,3,1,1,1,1,1,1,false,false>
      <<<blocksFor(D3,H3,W3), 256, 0, stream>>>(
      actA, pk[10], sc[10], bi[10], mask3, actB, nullptr, D3,H3,W3, D3,H3,W3);
  // w_out : 64->128, kernel (3,1,1), stride (2,1,1), pad 0, f32 out, LDS 48KB
  conv_bn_relu_wmma<64,128,3,1,1,2,1,1,0,0,0,true,true>
      <<<blocksFor(D4,H4,W4), 256, 48 * 1024, stream>>>(
      actB, pk[11], sc[11], bi[11], mask4, nullptr, (float*)d_out,
      D3,H3,W3, D4,H4,W4);
}
